// AttentionAggregate_Weight_59751585022698
// MI455X (gfx1250) — matrix operand order repaired and checked by reading, not compiled, not run on hardware
//
#include <hip/hip_runtime.h>

// ---------------------------------------------------------------------------
// AttentionAggregate for MI455X (gfx1250, wave32).
//  * Algebraic collapse: W1/W2 projections fold into v1 = a_w@W1, v2 = a_w@W2
//    (precomputed per launch into d_ws) -> kernel is pure HBM streaming,
//    ~676 MB @ 23.3 TB/s  => ~29 us roofline floor.
//  * score matvec s2 = middle_key[32x128] @ v2 done with v_wmma_f32_16x16x32_f16
//    from an LDS-staged, bank-conflict-padded f16 tile (f32 accumulate).
//  * value aggregation: coalesced non-temporal float4 streaming + lane shuffles.
// ---------------------------------------------------------------------------

typedef float    f4_t  __attribute__((ext_vector_type(4)));
typedef _Float16 h4_t  __attribute__((ext_vector_type(4)));
typedef _Float16 h8_t  __attribute__((ext_vector_type(8)));
typedef _Float16 v16h  __attribute__((ext_vector_type(16)));
typedef float    v8f   __attribute__((ext_vector_type(8)));

#define N_NODES 20000
#define KNB     32
#define DIM     128
#define WPB     4      // waves (= nodes) per block
#define ROWH    136    // padded LDS row length in halves (272 B: b128 reads conflict-free)

// ---- Kernel 0: v1[d] = sum_i a_w[i]*W1[i][d] ; v2[d] = sum_i a_w[i]*W2[i][d] ----
__global__ void AttnAgg_precompute_v(const float* __restrict__ W1,
                                     const float* __restrict__ W2,
                                     const float* __restrict__ aw,
                                     float* __restrict__ ws) {
  const int d = threadIdx.x;            // 128 threads, column-major reads are coalesced
  float a1 = 0.f, a2 = 0.f;
  for (int i = 0; i < DIM; ++i) {
    const float a = aw[i];
    a1 = fmaf(a, W1[i * DIM + d], a1);
    a2 = fmaf(a, W2[i * DIM + d], a2);
  }
  ws[d]       = a1;
  ws[DIM + d] = a2;
}

// ---- Main kernel: one wave32 per node ----
__global__ __launch_bounds__(WPB * 32)
void AttnAgg_main(const float* __restrict__ MK,   // [N,K,D] middle_key
                  const float* __restrict__ NK,   // [N,D]   nodes_key
                  const float* __restrict__ MV,   // [N,K,D] middle_value
                  const float* __restrict__ ws,   // v1 | v2
                  const float* __restrict__ ab,   // bias (1)
                  float* __restrict__ out) {      // [N,D]
  __shared__ __align__(16) float    v1s[DIM];
  __shared__ __align__(16) _Float16 v2s[DIM];
  __shared__ __align__(16) _Float16 tile[WPB][KNB][ROWH];
  __shared__ __align__(16) float    sbuf[WPB][KNB];

  const int t = threadIdx.x;
  if (t < DIM) {
    v1s[t] = ws[t];
    v2s[t] = (_Float16)ws[DIM + t];
  }
  __syncthreads();

  const int wv   = t >> 5;
  const int lane = t & 31;
  const int node = blockIdx.x * WPB + wv;
  if (node >= N_NODES) return;   // wave-uniform: EXEC stays all-ones for WMMA

  const float* mk = MK + (size_t)node * (KNB * DIM);
  const float* mv = MV + (size_t)node * (KNB * DIM);

  // ---- s1 = nodes_key[node] . v1  (lane covers 4 consecutive d) ----
  const f4_t nk = __builtin_nontemporal_load((const f4_t*)(NK + (size_t)node * DIM + lane * 4));
  float s1 = nk.x * v1s[lane * 4 + 0] + nk.y * v1s[lane * 4 + 1]
           + nk.z * v1s[lane * 4 + 2] + nk.w * v1s[lane * 4 + 3];
  #pragma unroll
  for (int m = 16; m > 0; m >>= 1) s1 += __shfl_xor(s1, m, 32);

  // ---- stage middle_key tile -> LDS as f16, fully coalesced (512B/instr) ----
  #pragma unroll 4
  for (int r = 0; r < KNB; ++r) {
    f4_t kv = __builtin_nontemporal_load((const f4_t*)(mk + r * DIM + lane * 4));
    h4_t hv;
    hv.x = (_Float16)kv.x; hv.y = (_Float16)kv.y;
    hv.z = (_Float16)kv.z; hv.w = (_Float16)kv.w;
    *(h4_t*)&tile[wv][r][lane * 4] = hv;
  }

  // ---- s2 = MK(32x128) @ v2 via V_WMMA_F32_16X16X32_F16 ----
  // A layout (16-bit, 16x32): lane l -> row M=l%16; v0..3 = K 8h..8h+7, v4..7 = K 16+8h..+7 (h=l/16)
  // B layout (32x16): same K order as A (v0..3 = K 8h..8h+7, v4..7 = K 16+8h..+7); only column 0 (lanes 0,16) carries v2
  const int m0 = lane & 15;
  const int hh = lane >> 4;
  v8f c0 = {};   // rows 0..15
  v8f c1 = {};   // rows 16..31
  const v16h bz = {};
  #pragma unroll
  for (int cc = 0; cc < DIM; cc += 32) {
    h8_t blo = *(const h8_t*)&v2s[cc + 8 * hh];
    h8_t bhi = *(const h8_t*)&v2s[cc + 16 + 8 * hh];
    v16h bf  = __builtin_shufflevector(blo, bhi, 0,1,2,3,4,5,6,7,8,9,10,11,12,13,14,15);
    bf = (m0 == 0) ? bf : bz;

    h8_t a0lo = *(const h8_t*)&tile[wv][m0][cc + 8 * hh];
    h8_t a0hi = *(const h8_t*)&tile[wv][m0][cc + 16 + 8 * hh];
    v16h a0   = __builtin_shufflevector(a0lo, a0hi, 0,1,2,3,4,5,6,7,8,9,10,11,12,13,14,15);
    h8_t a1lo = *(const h8_t*)&tile[wv][m0 + 16][cc + 8 * hh];
    h8_t a1hi = *(const h8_t*)&tile[wv][m0 + 16][cc + 16 + 8 * hh];
    v16h a1   = __builtin_shufflevector(a1lo, a1hi, 0,1,2,3,4,5,6,7,8,9,10,11,12,13,14,15);

    c0 = __builtin_amdgcn_wmma_f32_16x16x32_f16(false, a0, false, bf, (short)0, c0, false, false);
    c1 = __builtin_amdgcn_wmma_f32_16x16x32_f16(false, a1, false, bf, (short)0, c1, false, false);
  }

  // D column N=0 lives in lane 0 (M=0..7 across v0..7) and lane 16 (M=8..15).
  if (lane == 0) {
    #pragma unroll
    for (int r = 0; r < 8; ++r) { sbuf[wv][r]      = c0[r]; sbuf[wv][16 + r] = c1[r]; }
  } else if (lane == 16) {
    #pragma unroll
    for (int r = 0; r < 8; ++r) { sbuf[wv][8 + r]  = c0[r]; sbuf[wv][24 + r] = c1[r]; }
  }
  const float s2 = sbuf[wv][lane];   // same-wave LDS: in-order, compiler inserts dscnt wait

  // ---- score + softmax over K=32 (one score per lane) ----
  const float sc = tanhf(s1 + s2 + ab[0]);
  float mx = sc;
  #pragma unroll
  for (int m = 16; m > 0; m >>= 1) mx = fmaxf(mx, __shfl_xor(mx, m, 32));
  const float e = __expf(sc - mx);
  float sum = e;
  #pragma unroll
  for (int m = 16; m > 0; m >>= 1) sum += __shfl_xor(sum, m, 32);
  const float coef = e / sum;

  // ---- out[node] = coef @ middle_value[node] ; coalesced float4 stream ----
  f4_t acc = {};
  #pragma unroll 4
  for (int k = 0; k < KNB; ++k) {
    const float wk = __shfl(coef, k, 32);
    f4_t vv = __builtin_nontemporal_load((const f4_t*)(mv + k * DIM + lane * 4));
    acc.x = fmaf(wk, vv.x, acc.x);
    acc.y = fmaf(wk, vv.y, acc.y);
    acc.z = fmaf(wk, vv.z, acc.z);
    acc.w = fmaf(wk, vv.w, acc.w);
  }
  __builtin_nontemporal_store(acc, (f4_t*)(out + (size_t)node * DIM + lane * 4));
}

extern "C" void kernel_launch(void* const* d_in, const int* in_sizes, int n_in,
                              void* d_out, int out_size, void* d_ws, size_t ws_size,
                              hipStream_t stream) {
  const float* MK = (const float*)d_in[0];  // middle_key  [N,K,D]
  const float* NK = (const float*)d_in[1];  // nodes_key   [N,D]
  const float* MV = (const float*)d_in[2];  // middle_value[N,K,D]
  const float* W1 = (const float*)d_in[3];  // [D,D]
  const float* W2 = (const float*)d_in[4];  // [D,D]
  const float* AW = (const float*)d_in[5];  // [D]
  const float* AB = (const float*)d_in[6];  // [1]
  float* out = (float*)d_out;
  float* ws  = (float*)d_ws;                // 256 floats: v1 | v2

  AttnAgg_precompute_v<<<1, DIM, 0, stream>>>(W1, W2, AW, ws);

  const int blocks = (N_NODES + WPB - 1) / WPB;   // 5000
  AttnAgg_main<<<blocks, WPB * 32, 0, stream>>>(MK, NK, MV, ws, AB, out);
}
